// DGNNLayer_89996744720819
// MI455X (gfx1250) — compile-verified, matrix operand order checked by reading, not run on hardware
//
#include <hip/hip_runtime.h>
#include <hip/hip_bf16.h>
#include <math.h>

// Problem constants (fixed by the reference architecture; D must be 128)
#define DD  128
#define NHH 8

typedef __attribute__((ext_vector_type(8)))  float  v8f;
typedef __attribute__((ext_vector_type(16))) __bf16 v16bf;

union FragU { v16bf v; uint4 q[2]; };

__device__ __forceinline__ unsigned short f2bf(float f) {
  unsigned int u = __float_as_uint(f);
  if ((u & 0x7fffffffu) > 0x7f800000u) return (unsigned short)((u >> 16) | 0x40u); // quiet NaN
  u += 0x7fffu + ((u >> 16) & 1u);                                                 // RNE
  return (unsigned short)(u >> 16);
}

// A-fragment (16x32 bf16) from row-major LDS tile, per ISA 16-bit A layout:
// lane<16 holds K {k0..k0+7, k0+16..k0+23}; lane>=16 holds K {k0+8..k0+15, k0+24..k0+31}
__device__ __forceinline__ v16bf lds_a_frag(const unsigned short* base, int stride, int k0, int lane) {
  int row = lane & 15;
  int kb  = (lane < 16) ? 0 : 8;
  const unsigned short* p = base + row * stride + k0 + kb;
  FragU u;
  u.q[0] = *(const uint4*)(p);
  u.q[1] = *(const uint4*)(p + 16);
  return u.v;
}

// B-fragment from pre-packed weights: 32B contiguous per lane per (kt,nt)
__device__ __forceinline__ v16bf w_frag(const unsigned short* wp, int kt, int nt, int ntn, int lane) {
  const uint4* p = (const uint4*)(wp + ((((long)kt * ntn + nt) * 32 + lane) << 4));
  FragU u;
  u.q[0] = p[0];
  u.q[1] = p[1];
  return u.v;
}

#define WMMA_BF16(a, b, c) \
  __builtin_amdgcn_wmma_f32_16x16x32_bf16(false, (a), false, (b), (short)0, (c), false, false)

// ---------------------------------------------------------------- prep kernels

__global__ void cvt_bf16_kernel(const float* __restrict__ src, unsigned short* __restrict__ dst, long n) {
  long i = (long)blockIdx.x * blockDim.x + threadIdx.x;
  if (i < n) dst[i] = f2bf(src[i]);
}

__global__ void fill0_kernel(float* __restrict__ p, long n) {
  long i = (long)blockIdx.x * blockDim.x + threadIdx.x;
  if (i < n) p[i] = 0.0f;
}

// rvec[d] = rte_table[d] @ rte_W + rte_b   for d = 0,1,2
__global__ void rvec_kernel(const float* __restrict__ tab, const float* __restrict__ rte_W,
                            const float* __restrict__ rte_b, float* __restrict__ rvec) {
  int d = blockIdx.x, n = threadIdx.x;
  float s = 0.0f;
  #pragma unroll 8
  for (int k = 0; k < DD; k++) s += tab[d * DD + k] * rte_W[k * DD + n];
  rvec[d * DD + n] = s + rte_b[n];
}

// Repack f32 weight W[K x N] into bf16 per-lane WMMA B-fragments.
// dst flat index == (((kt*ntn + nt)*32 + lane)*16 + j)  which equals tid.
__global__ void pack_w_kernel(const float* __restrict__ src, unsigned short* __restrict__ dst,
                              int K, int N) {
  int tid = blockIdx.x * blockDim.x + threadIdx.x;
  if (tid >= K * N) return;
  int j    = tid & 15;
  int lane = (tid >> 4) & 31;
  int ntn  = N >> 4;
  int nt   = (tid >> 9) % ntn;
  int kt   = tid / (512 * ntn);
  int kb   = (lane < 16) ? 0 : 8;
  int k    = kt * 32 + ((j < 8) ? (kb + j) : (16 + kb + (j - 8)));
  int n    = nt * 16 + (lane & 15);
  dst[tid] = f2bf(src[k * N + n]);
}

// ---------------------------------------------------------------- q per node
// qn[node] = (x[tgt][node] + rvec0) @ q_W + q_b     (N x 128)
__global__ void q_node_kernel(const float* __restrict__ x_t, const float* __restrict__ rvec0,
                              const unsigned short* __restrict__ qWp, const float* __restrict__ q_b,
                              float* __restrict__ qn, int N) {
  __shared__ unsigned short sm[4 * 2048];           // 4 waves * 16x128 bf16
  int wave = threadIdx.x >> 5, lane = threadIdx.x & 31;
  int tile = blockIdx.x * 4 + wave;
  if (tile >= (N >> 4)) return;                     // wave-uniform
  unsigned short* A = sm + wave * 2048;
  int row = lane & 15, hc = lane >> 4;
  {
    const float* xr = x_t + ((long)(tile * 16 + row)) * DD + hc * 64;
    const float* rv = rvec0 + hc * 64;
    unsigned int* d = (unsigned int*)(A + row * 128 + hc * 64);
    #pragma unroll
    for (int j = 0; j < 64; j += 2) {
      unsigned int lo = f2bf(xr[j] + rv[j]);
      unsigned int hi = f2bf(xr[j + 1] + rv[j + 1]);
      d[j >> 1] = lo | (hi << 16);
    }
  }
  __builtin_amdgcn_wave_barrier();
  int col = lane & 15, half = lane >> 4;
  #pragma unroll
  for (int nt = 0; nt < 8; nt++) {
    v8f c = {};
    #pragma unroll
    for (int ks = 0; ks < 4; ks++)
      c = WMMA_BF16(lds_a_frag(A, 128, ks * 32, lane), w_frag(qWp, ks, nt, 8, lane), c);
    int g = nt * 16 + col;
    float qb = q_b[g];
    #pragma unroll
    for (int r = 0; r < 8; r++)
      qn[((long)(tile * 16 + r + 8 * half)) * DD + g] = c[r] + qb;
  }
}

// ---------------------------------------------------------------- edge kernel
// For one (tgt, s) block: msg = [x_src|attr]@msg_W + b + rvec[dt]  (bf16)
// k = msg@k_W + k_b ; att = <q[dst], k>/4 ; e = exp(att)
// denom[dst,h] += e ; emb[dst,:] += e * (msg@v_W + v_b)
__global__ void kv_edge_kernel(const unsigned short* __restrict__ xbf,
                               const unsigned short* __restrict__ abf,
                               const int* __restrict__ edge_index,
                               const unsigned short* __restrict__ msgWp,
                               const unsigned short* __restrict__ kWp,
                               const unsigned short* __restrict__ vWp,
                               const float* __restrict__ msg_b, const float* __restrict__ k_b,
                               const float* __restrict__ v_b,  const float* __restrict__ rvec,
                               const float* __restrict__ qn,
                               float* __restrict__ denom, float* __restrict__ emb,
                               int s, int dt, int E, int N) {
  __shared__ unsigned short sm[4 * 4096];           // 4 waves * 16x256 bf16 (reused for smv)
  int wave = threadIdx.x >> 5, lane = threadIdx.x & 31;
  int tile = blockIdx.x * 4 + wave;
  if (tile >= (E >> 4)) return;                     // wave-uniform
  unsigned short* Asrc = sm + wave * 4096;
  const int* srcn = edge_index + (long)s * 3 * E;
  const int* dstn = srcn + E;
  const int* aidx = srcn + 2 * E;
  int ebase = tile * 16;

  // stage A = [x_src(128) | attr(128)] bf16, 16 edges
  {
    int row = lane & 15, part = lane >> 4;
    int e = ebase + row;
    const unsigned short* src = (part == 0)
        ? (xbf + ((long)s * N + srcn[e]) * DD)
        : (abf + (long)aidx[e] * DD);
    const uint4* s4 = (const uint4*)src;
    uint4* d4 = (uint4*)(Asrc + row * 256 + part * 128);
    #pragma unroll
    for (int i = 0; i < 16; i++) d4[i] = s4[i];
  }
  __builtin_amdgcn_wave_barrier();

  int col = lane & 15, half = lane >> 4;

  // msg GEMM: K=256, Nout=128 (8 n-tiles kept in registers)
  v8f macc[8];
  #pragma unroll
  for (int nt = 0; nt < 8; nt++) {
    v8f c = {};
    #pragma unroll
    for (int ks = 0; ks < 8; ks++)
      c = WMMA_BF16(lds_a_frag(Asrc, 256, ks * 32, lane), w_frag(msgWp, ks, nt, 8, lane), c);
    macc[nt] = c;
  }
  __builtin_amdgcn_wave_barrier();

  // smv = msg + msg_b + rvec[dt], bf16, overwrite staging (GEMM done)
  unsigned short* Ssmv = Asrc;                       // 16x128 bf16
  #pragma unroll
  for (int nt = 0; nt < 8; nt++) {
    int g = nt * 16 + col;
    float bb = msg_b[g] + rvec[dt * DD + g];
    #pragma unroll
    for (int r = 0; r < 8; r++)
      Ssmv[(r + 8 * half) * 128 + g] = f2bf(macc[nt][r] + bb);
  }
  __builtin_amdgcn_wave_barrier();

  int dstR[8];
  #pragma unroll
  for (int r = 0; r < 8; r++) dstR[r] = dstn[ebase + r + 8 * half];

  // per head (== per 16-col n-tile): k GEMM, att, exp, v GEMM, scatter
  #pragma unroll
  for (int nt = 0; nt < NHH; nt++) {
    int g = nt * 16 + col;
    v8f kc = {};
    #pragma unroll
    for (int ks = 0; ks < 4; ks++)
      kc = WMMA_BF16(lds_a_frag(Ssmv, 128, ks * 32, lane), w_frag(kWp, ks, nt, 8, lane), kc);
    float kb = k_b[g];
    float tot[8];
    #pragma unroll
    for (int r = 0; r < 8; r++)
      tot[r] = (kc[r] + kb) * qn[(long)dstR[r] * DD + g];
    #pragma unroll
    for (int m = 1; m < 16; m <<= 1) {
      #pragma unroll
      for (int r = 0; r < 8; r++) tot[r] += __shfl_xor(tot[r], m, 32);
    }
    float ex[8];
    #pragma unroll
    for (int r = 0; r < 8; r++) ex[r] = expf(tot[r] * 0.25f);   // /sqrt(DK)=1/4; shift-free softmax
    if (col == 0) {
      #pragma unroll
      for (int r = 0; r < 8; r++) atomicAdd(&denom[(long)dstR[r] * NHH + nt], ex[r]);
    }
    v8f vc = {};
    #pragma unroll
    for (int ks = 0; ks < 4; ks++)
      vc = WMMA_BF16(lds_a_frag(Ssmv, 128, ks * 32, lane), w_frag(vWp, ks, nt, 8, lane), vc);
    float vb = v_b[g];
    #pragma unroll
    for (int r = 0; r < 8; r++)
      atomicAdd(&emb[(long)dstR[r] * DD + g], ex[r] * (vc[r] + vb));
  }
}

// ---------------------------------------------------------------- FFN kernel
// h = emb/denom + x[tgt]; LN; GELU-MLP 128->256->128; out = h + mlp
__global__ void ffn_kernel(const float* __restrict__ x_t, const float* __restrict__ emb,
                           const float* __restrict__ denom,
                           const float* __restrict__ ln_g, const float* __restrict__ ln_b,
                           const unsigned short* __restrict__ w1p, const float* __restrict__ b1,
                           const unsigned short* __restrict__ w2p, const float* __restrict__ b2,
                           float* __restrict__ out_t, int N) {
  __shared__ unsigned short sm[4 * 6144];            // per wave: A1 16x128 + act 16x256
  int wave = threadIdx.x >> 5, lane = threadIdx.x & 31;
  int tile = blockIdx.x * 4 + wave;
  if (tile >= (N >> 4)) return;                      // wave-uniform
  unsigned short* A1  = sm + wave * 6144;
  unsigned short* act = A1 + 2048;
  int row = lane & 15, hc = lane >> 4;
  long node = (long)tile * 16 + row;

  // h, LayerNorm stats (lane pair covers one row: cols hc*64..hc*64+63)
  float dns[4];
  #pragma unroll
  for (int t4 = 0; t4 < 4; t4++) {
    float dn = denom[node * NHH + hc * 4 + t4];
    dns[t4] = (dn > 0.0f) ? (1.0f / dn) : 0.0f;
  }
  const float* er = emb + node * DD + hc * 64;
  const float* xr = x_t + node * DD + hc * 64;
  float hv[64], sum = 0.0f, sq = 0.0f;
  #pragma unroll
  for (int j = 0; j < 64; j++) {
    float h = er[j] * dns[j >> 4] + xr[j];
    hv[j] = h; sum += h; sq += h * h;
  }
  sum += __shfl_xor(sum, 16, 32);
  sq  += __shfl_xor(sq, 16, 32);
  float mean = sum * (1.0f / DD);
  float var  = sq * (1.0f / DD) - mean * mean;
  float rs   = rsqrtf(var + 1e-5f);
  {
    unsigned int* d = (unsigned int*)(A1 + row * 128 + hc * 64);
    #pragma unroll
    for (int j = 0; j < 64; j += 2) {
      int c0 = hc * 64 + j;
      unsigned int lo = f2bf((hv[j]     - mean) * rs * ln_g[c0]     + ln_b[c0]);
      unsigned int hi = f2bf((hv[j + 1] - mean) * rs * ln_g[c0 + 1] + ln_b[c0 + 1]);
      d[j >> 1] = lo | (hi << 16);
    }
  }
  __builtin_amdgcn_wave_barrier();

  int col = lane & 15, half = lane >> 4;
  // GEMM1: 128 -> 256, exact GELU, bf16 into act
  #pragma unroll
  for (int nt = 0; nt < 16; nt++) {
    v8f c = {};
    #pragma unroll
    for (int ks = 0; ks < 4; ks++)
      c = WMMA_BF16(lds_a_frag(A1, 128, ks * 32, lane), w_frag(w1p, ks, nt, 16, lane), c);
    int g = nt * 16 + col;
    float bb = b1[g];
    #pragma unroll
    for (int r = 0; r < 8; r++) {
      float z = c[r] + bb;
      float ge = 0.5f * z * (1.0f + erff(z * 0.70710678118654752f));
      act[(r + 8 * half) * 256 + g] = f2bf(ge);
    }
  }
  __builtin_amdgcn_wave_barrier();

  // GEMM2: 256 -> 128, + residual h, write out
  #pragma unroll
  for (int nt = 0; nt < 8; nt++) {
    v8f c = {};
    #pragma unroll
    for (int ks = 0; ks < 8; ks++)
      c = WMMA_BF16(lds_a_frag(act, 256, ks * 32, lane), w_frag(w2p, ks, nt, 8, lane), c);
    int g = nt * 16 + col;
    float bb = b2[g];
    #pragma unroll
    for (int r = 0; r < 8; r++) {
      long nrow = (long)tile * 16 + r + 8 * half;
      float dn = denom[nrow * NHH + nt];
      float sc = (dn > 0.0f) ? (1.0f / dn) : 0.0f;
      float h = emb[nrow * DD + g] * sc + x_t[nrow * DD + g];
      out_t[nrow * DD + g] = h + c[r] + bb;
    }
  }
}

// ---------------------------------------------------------------- launcher

extern "C" void kernel_launch(void* const* d_in, const int* in_sizes, int n_in,
                              void* d_out, int out_size, void* d_ws, size_t ws_size,
                              hipStream_t stream) {
  const float* x         = (const float*)d_in[0];
  const float* edge_attr = (const float*)d_in[1];
  const float* msg_W = (const float*)d_in[2];
  const float* msg_b = (const float*)d_in[3];
  const float* q_W   = (const float*)d_in[4];
  const float* q_b   = (const float*)d_in[5];
  const float* k_W   = (const float*)d_in[6];
  const float* k_b   = (const float*)d_in[7];
  const float* v_W   = (const float*)d_in[8];
  const float* v_b   = (const float*)d_in[9];
  const float* ln_g  = (const float*)d_in[10];
  const float* ln_b  = (const float*)d_in[11];
  const float* rte_table = (const float*)d_in[12];
  const float* rte_W     = (const float*)d_in[13];
  const float* rte_b     = (const float*)d_in[14];
  const float* mlp_W1 = (const float*)d_in[15];
  const float* mlp_b1 = (const float*)d_in[16];
  const float* mlp_W2 = (const float*)d_in[17];
  const float* mlp_b2 = (const float*)d_in[18];
  const int*   eidx   = (const int*)d_in[19];

  const int T = in_sizes[20];                 // 4
  const int N = in_sizes[1] / DD;             // 50000
  const int E = in_sizes[19] / (3 * T);       // 150000
  float* out = (float*)d_out;

  // workspace carve (all chunks multiple of 256 B)
  char* w = (char*)d_ws;
  unsigned short* xbf = (unsigned short*)w;  w += (size_t)T * N * DD * 2;
  unsigned short* abf = (unsigned short*)w;  w += (size_t)N * DD * 2;
  float* rvec = (float*)w;                   w += 3 * DD * sizeof(float) + 256 - (3 * DD * 4) % 256;
  unsigned short* msgWp = (unsigned short*)w; w += 2 * DD * DD * 2;
  unsigned short* qWp   = (unsigned short*)w; w += DD * DD * 2;
  unsigned short* kWp   = (unsigned short*)w; w += DD * DD * 2;
  unsigned short* vWp   = (unsigned short*)w; w += DD * DD * 2;
  unsigned short* w1p   = (unsigned short*)w; w += DD * 2 * DD * 2;
  unsigned short* w2p   = (unsigned short*)w; w += 2 * DD * DD * 2;
  float* qn    = (float*)w;                  w += (size_t)N * DD * 4;
  float* denom = (float*)w;                  w += (size_t)N * NHH * 4;   // denom directly before emb
  float* emb   = (float*)w;                  w += (size_t)N * DD * 4;

  // ---- prep (every call; deterministic) ----
  {
    long nX = (long)T * N * DD;
    cvt_bf16_kernel<<<(unsigned)((nX + 255) / 256), 256, 0, stream>>>(x, xbf, nX);
    long nA = (long)N * DD;
    cvt_bf16_kernel<<<(unsigned)((nA + 255) / 256), 256, 0, stream>>>(edge_attr, abf, nA);
    rvec_kernel<<<3, DD, 0, stream>>>(rte_table, rte_W, rte_b, rvec);
    pack_w_kernel<<<(2 * DD * DD) / 256, 256, 0, stream>>>(msg_W, msgWp, 2 * DD, DD);
    pack_w_kernel<<<(DD * DD) / 256, 256, 0, stream>>>(q_W, qWp, DD, DD);
    pack_w_kernel<<<(DD * DD) / 256, 256, 0, stream>>>(k_W, kWp, DD, DD);
    pack_w_kernel<<<(DD * DD) / 256, 256, 0, stream>>>(v_W, vWp, DD, DD);
    pack_w_kernel<<<(DD * 2 * DD) / 256, 256, 0, stream>>>(mlp_W1, w1p, DD, 2 * DD);
    pack_w_kernel<<<(2 * DD * DD) / 256, 256, 0, stream>>>(mlp_W2, w2p, 2 * DD, DD);
  }

  const int tilesE = E / 16, tilesN = N / 16;
  const int blkE = (tilesE + 3) / 4, blkN = (tilesN + 3) / 4;

  for (int tgt = 0; tgt < T; tgt++) {
    long nz = (long)N * (NHH + DD);           // denom + emb contiguous
    fill0_kernel<<<(unsigned)((nz + 255) / 256), 256, 0, stream>>>(denom, nz);
    q_node_kernel<<<blkN, 128, 0, stream>>>(x + (long)tgt * N * DD, rvec, qWp, q_b, qn, N);
    int s0 = (tgt - 2 > 0) ? (tgt - 2) : 0;
    for (int s = s0; s <= tgt; s++) {
      kv_edge_kernel<<<blkE, 128, 0, stream>>>(xbf, abf, eidx, msgWp, kWp, vWp,
                                               msg_b, k_b, v_b, rvec, qn, denom, emb,
                                               s, tgt - s, E, N);
    }
    ffn_kernel<<<blkN, 128, 0, stream>>>(x + (long)tgt * N * DD, emb, denom, ln_g, ln_b,
                                         w1p, mlp_b1, w2p, mlp_b2,
                                         out + (long)tgt * N * DD, N);
  }
}
